// XLMRobertaSelfAttention_27779848470702
// MI455X (gfx1250) — matrix operand order repaired and checked by reading, not compiled, not run on hardware
//
#include <hip/hip_runtime.h>
#include <hip/hip_bf16.h>

typedef __bf16 bf16_t;
typedef __attribute__((ext_vector_type(16))) __bf16 v16bf;
typedef __attribute__((ext_vector_type(8)))  float  v8f;

#define BATCH     4
#define SEQ       1024
#define HID       1024
#define NHEADS    16
#define HDIM      64
#define EROWS     2048   // 2047 real rows + 1 zero pad row

union FragB16 {
    v16bf  v;
    bf16_t h[16];
    uint4  u4[2];
};

__device__ __forceinline__ v8f wmma_bf16(v16bf a, v16bf b, v8f c) {
    return __builtin_amdgcn_wmma_f32_16x16x32_bf16(
        /*neg_a=*/false, a, /*neg_b=*/false, b,
        /*c_mod=*/(short)0, c, /*reuse_a=*/false, /*reuse_b=*/false);
}

// ---------------------------------------------------------------------------
// Kernel 0: convert dist_emb (2047x64 f32) -> bf16, padded to 2048 rows.
// ---------------------------------------------------------------------------
__global__ void __launch_bounds__(256) cvt_emb_kernel(const float* __restrict__ E,
                                                      bf16_t* __restrict__ Eb) {
    int i = blockIdx.x * 256 + threadIdx.x;
    if (i < EROWS * HDIM) {
        float v = (i < 2047 * HDIM) ? E[i] : 0.0f;
        Eb[i] = (bf16_t)v;
    }
}

// ---------------------------------------------------------------------------
// Kernel 1: fused QKV projection, X[4096,1024] @ W^T[1024,3072] + b.
// 256 threads = 8 waves; block tile 64(M) x 64(N); each wave: 2 C tiles.
// Writes Qb/Kb as [b,h,s,d] bf16 and V transposed as Vt [b,h,d,s] bf16.
// ---------------------------------------------------------------------------
__device__ __forceinline__ void cvt8(FragB16& f, int off, float4 a, float4 b) {
    f.h[off+0] = (bf16_t)a.x; f.h[off+1] = (bf16_t)a.y;
    f.h[off+2] = (bf16_t)a.z; f.h[off+3] = (bf16_t)a.w;
    f.h[off+4] = (bf16_t)b.x; f.h[off+5] = (bf16_t)b.y;
    f.h[off+6] = (bf16_t)b.z; f.h[off+7] = (bf16_t)b.w;
}

__global__ void __launch_bounds__(256) qkv_proj_kernel(
        const float* __restrict__ X, const float* __restrict__ W,
        const float* __restrict__ bias,
        bf16_t* __restrict__ Qb, bf16_t* __restrict__ Kb, bf16_t* __restrict__ Vt) {
    const int lane = threadIdx.x & 31;
    const int wid  = threadIdx.x >> 5;
    const int l16  = lane & 15;
    const int half = lane >> 4;

    const int mBase = blockIdx.x * 64;
    const int nBase = blockIdx.y * 64;
    const int nt  = wid & 3;            // N sub-tile (shared by both C tiles)
    const int mt0 = wid >> 2;           // 0..1
    const int mt1 = mt0 + 2;            // 2..3

    const int colLocal = nt * 16 + l16; // 0..63 within N block
    const int oCol     = nBase + colLocal;

    const float* wrow  = W + (size_t)oCol * HID;
    const float* xrow0 = X + (size_t)(mBase + mt0 * 16 + l16) * HID;
    const float* xrow1 = X + (size_t)(mBase + mt1 * 16 + l16) * HID;

    v8f acc0 = {}; v8f acc1 = {};
    for (int k0 = 0; k0 < HID; k0 += 32) {
        // B fragment: W row, interleaved K layout (e<8: k0+half*8+e ; e>=8: k0+16+half*8+e-8)
        FragB16 bf;
        {
            const float4* p = (const float4*)(wrow + k0 + half * 8);
            const float4* q = (const float4*)(wrow + k0 + 16 + half * 8);
            cvt8(bf, 0, p[0], p[1]);
            cvt8(bf, 8, q[0], q[1]);
        }
        // A fragments: interleaved K layout (e<8: k0+half*8+e ; e>=8: k0+16+half*8+e-8)
        FragB16 a0, a1;
        {
            const float4* p = (const float4*)(xrow0 + k0 + half * 8);
            const float4* q = (const float4*)(xrow0 + k0 + 16 + half * 8);
            cvt8(a0, 0, p[0], p[1]);
            cvt8(a0, 8, q[0], q[1]);
        }
        {
            const float4* p = (const float4*)(xrow1 + k0 + half * 8);
            const float4* q = (const float4*)(xrow1 + k0 + 16 + half * 8);
            cvt8(a1, 0, p[0], p[1]);
            cvt8(a1, 8, q[0], q[1]);
        }
        acc0 = wmma_bf16(a0.v, bf.v, acc0);
        acc1 = wmma_bf16(a1.v, bf.v, acc1);
    }

    const float bval = bias[oCol];
    const int b      = mBase / SEQ;
    const int s0     = mBase % SEQ;
    const int region = nBase / HID;               // 0=q 1=k 2=v
    const int head   = (nBase % HID) / HDIM;
    const int d      = colLocal;                  // 0..63
    const size_t bh  = (size_t)(b * NHEADS + head);

    if (region < 2) {
        bf16_t* dst = (region == 0) ? Qb : Kb;
        #pragma unroll
        for (int j = 0; j < 8; ++j) {
            int s = s0 + mt0 * 16 + j + half * 8;
            dst[(bh * SEQ + s) * HDIM + d] = (bf16_t)(acc0[j] + bval);
        }
        #pragma unroll
        for (int j = 0; j < 8; ++j) {
            int s = s0 + mt1 * 16 + j + half * 8;
            dst[(bh * SEQ + s) * HDIM + d] = (bf16_t)(acc1[j] + bval);
        }
    } else {
        // V transposed store: Vt[b,h,d,s]; 8 consecutive s per lane -> one 16B store
        FragB16 pk;
        #pragma unroll
        for (int j = 0; j < 8; ++j) pk.h[j] = (bf16_t)(acc0[j] + bval);
        int sb = s0 + mt0 * 16 + half * 8;
        *(uint4*)(Vt + (bh * HDIM + d) * SEQ + sb) = pk.u4[0];
        #pragma unroll
        for (int j = 0; j < 8; ++j) pk.h[j] = (bf16_t)(acc1[j] + bval);
        sb = s0 + mt1 * 16 + half * 8;
        *(uint4*)(Vt + (bh * HDIM + d) * SEQ + sb) = pk.u4[0];
    }
}

// ---------------------------------------------------------------------------
// Kernel 2: flash attention with relative_key_query bias, all-WMMA.
// Grid: B*h*(S/64) = 1024 blocks, 128 threads = 4 independent waves.
// Wave owns 16 query rows; loops r in chunks of 32.
// ---------------------------------------------------------------------------
__global__ void __launch_bounds__(128) attn_kernel(
        const bf16_t* __restrict__ Qb, const bf16_t* __restrict__ Kb,
        const bf16_t* __restrict__ Vt, const bf16_t* __restrict__ Eb,
        const float* __restrict__ mask, float* __restrict__ out) {
    __shared__ float lds[4][1536];   // per wave: g1[512] | g2[512] | scores[512]

    const int lane = threadIdx.x & 31;
    const int wid  = threadIdx.x >> 5;
    const int l16  = lane & 15;
    const int half = lane >> 4;

    const int idx = blockIdx.x;
    const int b   = idx >> 8;          // / (16*16)
    const int h   = (idx >> 4) & 15;
    const int lb  = idx & 15;
    const int lw  = lb * 64 + wid * 16;   // query row base for this wave

    const size_t bh = (size_t)(b * NHEADS + h);
    const bf16_t* Qbh = Qb + bh * SEQ * HDIM;
    const bf16_t* Kbh = Kb + bh * SEQ * HDIM;
    const bf16_t* Vbh = Vt + bh * HDIM * SEQ;
    const float*  mkb = mask + (size_t)b * SEQ;

    float* g1 = lds[wid];
    float* g2 = lds[wid] + 512;
    float* sl = lds[wid] + 1024;

    // Q A-fragments, resident for whole kernel (row = lw + l16)
    FragB16 qf[2];
    {
        const bf16_t* qrow = Qbh + (size_t)(lw + l16) * HDIM;
        #pragma unroll
        for (int kc = 0; kc < 2; ++kc) {
            qf[kc].u4[0] = *(const uint4*)(qrow + kc * 32 + half * 16);
            qf[kc].u4[1] = *(const uint4*)(qrow + kc * 32 + half * 16 + 8);
        }
    }

    const float SCALE = 0.125f * 1.4426950408889634f;  // 1/sqrt(64) * log2(e)
    const float L2E   = 1.4426950408889634f;

    float mstat = -3.0e38f;
    float lstat = 0.0f;
    v8f oacc0 = {}, oacc1 = {}, oacc2 = {}, oacc3 = {};

    for (int r0 = 0; r0 < SEQ; r0 += 32) {
        #pragma unroll
        for (int tile = 0; tile < 2; ++tile) {
            const int rt = r0 + tile * 16;
            const bf16_t* krow = Kbh + (size_t)(rt + l16) * HDIM;

            // K fragments: B layout (QK^T) and A layout (for G2 = K @ E^T)
            FragB16 kB[2], kA[2];
            #pragma unroll
            for (int kc = 0; kc < 2; ++kc) {
                kB[kc].u4[0] = *(const uint4*)(krow + kc * 32 + half * 16);
                kB[kc].u4[1] = *(const uint4*)(krow + kc * 32 + half * 16 + 8);
                kA[kc].u4[0] = *(const uint4*)(krow + kc * 32 + half * 16);
                kA[kc].u4[1] = *(const uint4*)(krow + kc * 32 + half * 16 + 8);
            }

            // raw scores: Q(16x64) @ K^T(64x16)
            v8f sacc = {};
            #pragma unroll
            for (int kc = 0; kc < 2; ++kc) sacc = wmma_bf16(qf[kc].v, kB[kc].v, sacc);

            // bias GEMMs against 32-row E window (rows c-15 .. c+16, c = lw-rt+1023)
            const int erowBase = lw - rt + 1008;   // = c - 15, in [0, 2016]
            #pragma unroll
            for (int nt = 0; nt < 2; ++nt) {
                v8f ga = {}; v8f gb = {};
                const bf16_t* erow = Eb + (size_t)(erowBase + nt * 16 + l16) * HDIM;
                #pragma unroll
                for (int kc = 0; kc < 2; ++kc) {
                    FragB16 ef;   // B layout: 16 consecutive d at kc*32 + half*16
                    ef.u4[0] = *(const uint4*)(erow + kc * 32 + half * 16);
                    ef.u4[1] = *(const uint4*)(erow + kc * 32 + half * 16 + 8);
                    ga = wmma_bf16(qf[kc].v, ef.v, ga);   // G1 = Q @ E^T
                    gb = wmma_bf16(kA[kc].v, ef.v, gb);   // G2 = K @ E^T
                }
                #pragma unroll
                for (int j = 0; j < 8; ++j) {             // C-layout scatter to LDS
                    int u = j + half * 8;
                    g1[u * 32 + nt * 16 + l16] = ga[j];
                    g2[u * 32 + nt * 16 + l16] = gb[j];
                }
            }
            asm volatile("" ::: "memory");  // order LDS write -> gather (DS is in-order per wave)

            // diagonal gather + scale + mask; stash scores in LDS (C-layout)
            const float mv = mkb[rt + l16] * L2E;
            #pragma unroll
            for (int j = 0; j < 8; ++j) {
                int u = j + half * 8;
                int t = u - l16 + 15;                     // in [0,30]
                float s = (sacc[j] + g1[u * 32 + t] + g2[l16 * 32 + t]) * SCALE + mv;
                sl[u * 32 + tile * 16 + l16] = s;
            }
            asm volatile("" ::: "memory");
        }

        // Transposed read-back: lane owns score row l16, cols half*16..+15.
        // This is exactly the B-fragment layout for P^T and makes stats per-lane.
        float sc[16];
        #pragma unroll
        for (int e = 0; e < 16; ++e) sc[e] = sl[l16 * 32 + half * 16 + e];

        float cmax = sc[0];
        #pragma unroll
        for (int e = 1; e < 16; ++e) cmax = fmaxf(cmax, sc[e]);
        float rmax  = fmaxf(cmax, __shfl_xor(cmax, 16, 32));
        float mnew  = fmaxf(mstat, rmax);
        float alpha = __builtin_exp2f(mstat - mnew);

        FragB16 pf;
        float csum = 0.0f;
        #pragma unroll
        for (int e = 0; e < 16; ++e) {
            float p = __builtin_exp2f(sc[e] - mnew);
            csum += p;
            pf.h[e] = (bf16_t)p;
        }
        float rsum = csum + __shfl_xor(csum, 16, 32);
        lstat = lstat * alpha + rsum;
        mstat = mnew;

        // O^T(64x16) += V^T(16x32 per dt) @ P^T(32x16); rescale is per-lane (col=l16)
        v8f* oaccs[4] = { &oacc0, &oacc1, &oacc2, &oacc3 };
        #pragma unroll
        for (int dt = 0; dt < 4; ++dt) {
            const bf16_t* vrow = Vbh + (size_t)(dt * 16 + l16) * SEQ + r0;
            FragB16 vf;   // A layout over K=r
            vf.u4[0] = *(const uint4*)(vrow + half * 16);
            vf.u4[1] = *(const uint4*)(vrow + half * 16 + 8);
            v8f o = *oaccs[dt];
            #pragma unroll
            for (int j = 0; j < 8; ++j) o[j] *= alpha;
            *oaccs[dt] = wmma_bf16(vf.v, pf.v, o);
        }

        if (r0 + 32 < SEQ) {   // global_prefetch_b8 for next chunk (L2 resident)
            __builtin_prefetch(Kbh + (size_t)(r0 + 32 + l16) * HDIM, 0, 3);
            __builtin_prefetch(Vbh + (size_t)l16 * SEQ + r0 + 32, 0, 3);
        }
    }

    // normalize and store: per lane two coalesced float4 per dt
    const float inv   = 1.0f / lstat;
    const int   s_tok = lw + l16;
    float* orow = out + ((size_t)(b * SEQ + s_tok)) * HID + h * HDIM;
    v8f* oaccs[4] = { &oacc0, &oacc1, &oacc2, &oacc3 };
    #pragma unroll
    for (int dt = 0; dt < 4; ++dt) {
        v8f o = *oaccs[dt];
        float4 lo, hi;
        lo.x = o[0] * inv; lo.y = o[1] * inv; lo.z = o[2] * inv; lo.w = o[3] * inv;
        hi.x = o[4] * inv; hi.y = o[5] * inv; hi.z = o[6] * inv; hi.w = o[7] * inv;
        float4* dst = (float4*)(orow + dt * 16 + half * 8);
        dst[0] = lo;
        dst[1] = hi;
    }
}

// ---------------------------------------------------------------------------
extern "C" void kernel_launch(void* const* d_in, const int* in_sizes, int n_in,
                              void* d_out, int out_size, void* d_ws, size_t ws_size,
                              hipStream_t stream) {
    (void)in_sizes; (void)n_in; (void)out_size; (void)ws_size;

    const float* hidden = (const float*)d_in[0];   // [4,1024,1024]
    const float* qkv_w  = (const float*)d_in[1];   // [3072,1024]
    const float* qkv_b  = (const float*)d_in[2];   // [3072]
    const float* diste  = (const float*)d_in[3];   // [2047,64]
    const float* emask  = (const float*)d_in[4];   // [4,1,1,1024]
    float*       outp   = (float*)d_out;           // [4,1024,1024]

    char* ws = (char*)d_ws;
    const size_t SZQKV = (size_t)BATCH * NHEADS * SEQ * HDIM * sizeof(bf16_t); // 8 MiB
    bf16_t* Qb = (bf16_t*)(ws);
    bf16_t* Kb = (bf16_t*)(ws + SZQKV);
    bf16_t* Vt = (bf16_t*)(ws + 2 * SZQKV);
    bf16_t* Eb = (bf16_t*)(ws + 3 * SZQKV);        // 2048*64 bf16 = 256 KiB

    cvt_emb_kernel<<<dim3((EROWS * HDIM + 255) / 256), dim3(256), 0, stream>>>(diste, Eb);
    qkv_proj_kernel<<<dim3(64, 48), dim3(256), 0, stream>>>(hidden, qkv_w, qkv_b, Qb, Kb, Vt);
    attn_kernel<<<dim3(BATCH * NHEADS * (SEQ / 64)), dim3(128), 0, stream>>>(
        Qb, Kb, Vt, Eb, emask, outp);
}
